// MultiTaskGNN_51092930953620
// MI455X (gfx1250) — matrix operand order repaired and checked by reading, not compiled, not run on hardware
//
#include <hip/hip_runtime.h>
#include <hip/hip_bf16.h>

typedef __attribute__((ext_vector_type(16))) _Float16 v16h;
typedef __attribute__((ext_vector_type(8)))  float    v8f;

#define HDIM 128
#define LDSK 136   // padded K-stride (halves) to dodge LDS bank conflicts

// ---------------------------------------------------------------- utility
__global__ void k_zero(float* __restrict__ p, int n) {
    int i = blockIdx.x * blockDim.x + threadIdx.x;
    if (i < n) p[i] = 0.0f;
}

__global__ void k_deg_init(float* __restrict__ deg, int n) {
    int i = blockIdx.x * blockDim.x + threadIdx.x;
    if (i < n) deg[i] = 1.0f;                 // self-loop contributes 1
}

__global__ void k_deg_accum(const long long* __restrict__ dst,
                            float* __restrict__ deg, int nE) {
    int e = blockIdx.x * blockDim.x + threadIdx.x;
    if (e < nE) unsafeAtomicAdd(&deg[(int)dst[e]], 1.0f);
}

__global__ void k_rsqrt(const float* __restrict__ deg,
                        float* __restrict__ dinv, int n) {
    int i = blockIdx.x * blockDim.x + threadIdx.x;
    if (i < n) dinv[i] = rsqrtf(deg[i]);
}

// pack per-edge record {src, dst, coef, 0} into one 16B load
__global__ void k_prep_edges(const long long* __restrict__ src,
                             const long long* __restrict__ dst,
                             const float* __restrict__ dinv,
                             int4* __restrict__ rec, int nE) {
    int e = blockIdx.x * blockDim.x + threadIdx.x;
    if (e >= nE) return;
    int s = (int)src[e], d = (int)dst[e];
    float c = dinv[s] * dinv[d];
    rec[e] = make_int4(s, d, __float_as_int(c), 0);
}

// ---------------------------------------------------------------- WMMA GEMM
// C[M x 128] = A[M x 128] @ W[128 x 128]  (+ bias, relu when FUSE)
// Block = 256 threads = 8 waves; each wave owns a 16-row M tile, all 8 col tiles.
// Out-of-range rows are clamped to row 0: their (garbage) outputs are never
// stored, and D rows depend only on the matching A row, so no masking needed.
template <bool FUSE>
__global__ void __launch_bounds__(256)
k_gemm128(const float* __restrict__ A, const float* __restrict__ W,
          const float* __restrict__ bias, float* __restrict__ C, int M) {
    // W staged in LDS, transposed to [n][k] in f16, padded stride
    __shared__ _Float16 sWt[HDIM * LDSK];

    int tid = threadIdx.x;
    for (int i = tid; i < HDIM * HDIM; i += 256) {
        int k = i >> 7, n = i & 127;
        sWt[n * LDSK + k] = (_Float16)W[i];
    }

    const int wave = tid >> 5;
    const int lane = tid & 31;
    const int hl   = lane >> 4;          // half-wave: 0 or 1
    const int l16  = lane & 15;

    const int m0   = blockIdx.x * 128 + wave * 16;
    const int mrow = m0 + l16;
    const float* arow = A + (size_t)(mrow < M ? mrow : 0) * HDIM;

    // ---- load ALL A fragments up front (16x b128), convert to f16 once.
    // 16-bit A 16x32 layout per k-block: lane needs K = kb+hl*8+[0..7] and
    // K = kb+16+hl*8+[0..7]: two contiguous 8-float runs.
    v16h afrag[4];
#pragma unroll
    for (int kb4 = 0; kb4 < 4; ++kb4) {
        const int kb = kb4 * 32;
        const float4* ap = (const float4*)(arow + kb + hl * 8);
        const float4* aq = (const float4*)(arow + kb + 16 + hl * 8);
        float4 p0 = ap[0], p1 = ap[1];
        float4 p2 = aq[0], p3 = aq[1];
        v16h af;
        af[0]  = (_Float16)p0.x; af[1]  = (_Float16)p0.y;
        af[2]  = (_Float16)p0.z; af[3]  = (_Float16)p0.w;
        af[4]  = (_Float16)p1.x; af[5]  = (_Float16)p1.y;
        af[6]  = (_Float16)p1.z; af[7]  = (_Float16)p1.w;
        af[8]  = (_Float16)p2.x; af[9]  = (_Float16)p2.y;
        af[10] = (_Float16)p2.z; af[11] = (_Float16)p2.w;
        af[12] = (_Float16)p3.x; af[13] = (_Float16)p3.y;
        af[14] = (_Float16)p3.z; af[15] = (_Float16)p3.w;
        afrag[kb4] = af;
    }

    __syncthreads();   // sWt ready

    v8f acc[8] = {};

#pragma unroll
    for (int kb4 = 0; kb4 < 4; ++kb4) {
        const int kb = kb4 * 32;
#pragma unroll
        for (int nt = 0; nt < 8; ++nt) {
            // B fragment: 32x16 layout (lane half hl: K=hl*8+[0..7], 16+hl*8+[0..7])
            const _Float16* bp = &sWt[(nt * 16 + l16) * LDSK + kb + hl * 8];
            v16h bf;
#pragma unroll
            for (int j = 0; j < 16; ++j) bf[j] = bp[(j & 7) + ((j >> 3) << 4)];
            acc[nt] = __builtin_amdgcn_wmma_f32_16x16x32_f16(
                false, afrag[kb4], false, bf, (short)0, acc[nt], false, false);
        }
    }

    // C layout: VGPR r -> M = m0 + hl*8 + r, N = nt*16 + l16
#pragma unroll
    for (int nt = 0; nt < 8; ++nt) {
        const int n = nt * 16 + l16;
        const float bn = FUSE ? bias[n] : 0.0f;
#pragma unroll
        for (int r = 0; r < 8; ++r) {
            int m = m0 + hl * 8 + r;
            if (m < M) {
                float v = acc[nt][r];
                if (FUSE) v = fmaxf(v + bn, 0.0f);
                C[(size_t)m * HDIM + n] = v;
            }
        }
    }
}

// ---------------------------------------------------------------- edge scatter
// 32 threads per edge, 4 floats each: b128 metadata + b128 gather
// + 4x global_atomic_add_f32
__global__ void k_scatter(const int4* __restrict__ rec,
                          const float* __restrict__ h,
                          float* __restrict__ agg, int nE) {
    int idx = blockIdx.x * blockDim.x + threadIdx.x;
    int e = idx >> 5;
    if (e >= nE) return;
    int c = (idx & 31) << 2;
    int4 r = rec[e];
    float coef = __int_as_float(r.z);
    const float4 v = *(const float4*)(h + (size_t)r.x * HDIM + c);
    float* o = agg + (size_t)r.y * HDIM + c;
    unsafeAtomicAdd(o + 0, v.x * coef);
    unsafeAtomicAdd(o + 1, v.y * coef);
    unsafeAtomicAdd(o + 2, v.z * coef);
    unsafeAtomicAdd(o + 3, v.w * coef);
}

// fold self-loop (hW * dinv^2) + bias + relu, in place on agg
__global__ void k_selfbias_relu(const float* __restrict__ hW,
                                const float* __restrict__ dinv,
                                const float* __restrict__ bias,
                                float* __restrict__ agg, int N) {
    int i = blockIdx.x * blockDim.x + threadIdx.x;
    if (i >= N * HDIM) return;
    int node = i >> 7, f = i & 127;
    float di = dinv[node];
    agg[i] = fmaxf(agg[i] + hW[i] * di * di + bias[f], 0.0f);
}

// ---------------------------------------------------------------- pooling
__global__ void k_pool_accum(const float* __restrict__ h,
                             const long long* __restrict__ batch,
                             float* __restrict__ sums, float* __restrict__ cnt,
                             int N) {
    int i = blockIdx.x * blockDim.x + threadIdx.x;
    if (i >= N * HDIM) return;
    int node = i >> 7, f = i & 127;
    int g = (int)batch[node];
    unsafeAtomicAdd(&sums[g * HDIM + f], h[i]);
    if (f == 0) unsafeAtomicAdd(&cnt[g], 1.0f);
}

__global__ void k_pool_div(float* __restrict__ sums, const float* __restrict__ cnt,
                           int G) {
    int i = blockIdx.x * blockDim.x + threadIdx.x;
    if (i >= G * HDIM) return;
    sums[i] = sums[i] / fmaxf(cnt[i >> 7], 1.0f);
}

// ---------------------------------------------------------------- task head
// out[t,g,c] = sum_h z[g,h] * Wh[t,h,c] + bh[t,c]
__global__ void k_head(const float* __restrict__ z, const float* __restrict__ Wh,
                       const float* __restrict__ bh, float* __restrict__ out,
                       int T, int G) {
    int idx = blockIdx.x * blockDim.x + threadIdx.x;
    if (idx >= T * G * 2) return;
    int t = idx / (G * 2);
    int g = (idx >> 1) % G;
    int c = idx & 1;
    float acc = bh[t * 2 + c];
    const float* zr = z + (size_t)g * HDIM;
    const float* wr = Wh + (size_t)t * HDIM * 2 + c;
#pragma unroll 8
    for (int h = 0; h < HDIM; ++h) acc += zr[h] * wr[h * 2];
    out[idx] = acc;
}

// ---------------------------------------------------------------- launcher
extern "C" void kernel_launch(void* const* d_in, const int* in_sizes, int n_in,
                              void* d_out, int out_size, void* d_ws, size_t ws_size,
                              hipStream_t stream) {
    const float*     x     = (const float*)d_in[0];
    const long long* ei    = (const long long*)d_in[1];
    const long long* batch = (const long long*)d_in[2];
    const float* W1  = (const float*)d_in[3];
    const float* b1  = (const float*)d_in[4];
    const float* W2  = (const float*)d_in[5];
    const float* b2  = (const float*)d_in[6];
    const float* Wfc = (const float*)d_in[7];
    const float* bfc = (const float*)d_in[8];
    const float* Wh  = (const float*)d_in[9];
    const float* bh  = (const float*)d_in[10];
    float* out = (float*)d_out;

    const int N = in_sizes[2];
    const int E = in_sizes[1] / 2;
    const int T = in_sizes[9] / (HDIM * 2);
    const int G = out_size / (2 * T);
    const long long* src = ei;
    const long long* dst = ei + E;

    // workspace carve-up (floats)
    float* ws = (float*)d_ws;
    size_t o = 0;
    float* deg    = ws + o; o += N;
    float* dinv   = ws + o; o += N;
    float* bufA   = ws + o; o += (size_t)N * HDIM;   // GEMM outputs
    float* bufB   = ws + o; o += (size_t)N * HDIM;   // agg / h
    float* pooled = ws + o; o += (size_t)G * HDIM;
    float* cnt    = ws + o; o += G;
    float* z      = ws + o; o += (size_t)G * HDIM;
    int4*  erec   = (int4*)(ws + o); o += (size_t)E * 4;
    (void)ws_size; (void)n_in;

    const int TB = 256;
    const int NH = N * HDIM;
    const int mBlocks = (N + 127) / 128;
    const int gBlocks = (G + 127) / 128;

    // degrees + normalization + edge records
    k_deg_init<<<(N + TB - 1) / TB, TB, 0, stream>>>(deg, N);
    k_deg_accum<<<(E + TB - 1) / TB, TB, 0, stream>>>(dst, deg, E);
    k_rsqrt<<<(N + TB - 1) / TB, TB, 0, stream>>>(deg, dinv, N);
    k_prep_edges<<<(E + TB - 1) / TB, TB, 0, stream>>>(src, dst, dinv, erec, E);

    // ---- layer 1
    k_gemm128<false><<<mBlocks, TB, 0, stream>>>(x, W1, nullptr, bufA, N);
    k_zero<<<(NH + TB - 1) / TB, TB, 0, stream>>>(bufB, NH);
    {
        long long threads = (long long)E * 32;
        k_scatter<<<(unsigned)((threads + TB - 1) / TB), TB, 0, stream>>>(
            erec, bufA, bufB, E);
    }
    k_selfbias_relu<<<(NH + TB - 1) / TB, TB, 0, stream>>>(bufA, dinv, b1, bufB, N);

    // ---- layer 2
    k_gemm128<false><<<mBlocks, TB, 0, stream>>>(bufB, W2, nullptr, bufA, N);
    k_zero<<<(NH + TB - 1) / TB, TB, 0, stream>>>(bufB, NH);
    {
        long long threads = (long long)E * 32;
        k_scatter<<<(unsigned)((threads + TB - 1) / TB), TB, 0, stream>>>(
            erec, bufA, bufB, E);
    }
    k_selfbias_relu<<<(NH + TB - 1) / TB, TB, 0, stream>>>(bufA, dinv, b2, bufB, N);

    // ---- mean pool
    k_zero<<<(G * HDIM + G + TB - 1) / TB, TB, 0, stream>>>(pooled, G * HDIM + G);
    k_pool_accum<<<(NH + TB - 1) / TB, TB, 0, stream>>>(bufB, batch, pooled, cnt, N);
    k_pool_div<<<(G * HDIM + TB - 1) / TB, TB, 0, stream>>>(pooled, cnt, G);

    // ---- FC + relu (WMMA, fused epilogue)
    k_gemm128<true><<<gBlocks, TB, 0, stream>>>(pooled, Wfc, bfc, z, G);

    // ---- multi-task head
    k_head<<<(T * G * 2 + TB - 1) / TB, TB, 0, stream>>>(z, Wh, bh, out, T, G);
}
